// E71MatrixGated_55456617726103
// MI455X (gfx1250) — matrix operand order repaired and checked by reading, not compiled, not run on hardware
//
// CDNA5 (gfx1250) gated matrix-state recurrent layer.
// Pipeline: split-f16 -> WMMA in_proj+SiLU -> WMMA kvqa proj -> VALU scan
//           (state rows in VGPRs, one WG per batch) -> WMMA out_proj.
// GEMMs: v_wmma_f32_16x16x32_f16 with split-precision (hi/lo f16) operands
// (hi*hi + hi*lo + lo*hi, f32 accumulate ~ fp32 accuracy), and TDM
// tensor_load_to_lds double-buffered staging (TENSORcnt + s_wait_tensorcnt).
// TDM descriptor pad feature (1 dword per 16) reproduces the [128][34]-half
// padded LDS tile layout so ds fragment gathers stay bank-conflict-free.
// NOTE: this toolchain exposes the 6-arg tensor_load_to_lds builtin
// (u32x4, i32x8, i32x4, i32x4, i32x8, i32 cpol).

#include <hip/hip_runtime.h>
#include <stdint.h>

typedef _Float16     v16h  __attribute__((ext_vector_type(16)));
typedef float        v8f   __attribute__((ext_vector_type(8)));
typedef unsigned int u32x4 __attribute__((ext_vector_type(4)));
typedef int          i32x8 __attribute__((ext_vector_type(8)));
typedef int          i32x4 __attribute__((ext_vector_type(4)));

#define DIM   1024
#define NST   64
#define BB    16
#define TT    2048
#define ROWS  (BB*TT)          // 32768 (b*T+t row index, matches x/out layout)

// ---------------------------------------------------------------- split
__global__ void k_split(const float* __restrict__ in,
                        _Float16* __restrict__ hi,
                        _Float16* __restrict__ lo, size_t n) {
  size_t i = (size_t)blockIdx.x * blockDim.x + threadIdx.x;
  size_t stride = (size_t)gridDim.x * blockDim.x;
  for (; i < n; i += stride) {
    float v = in[i];
    _Float16 h = (_Float16)v;
    hi[i] = h;
    lo[i] = (_Float16)(v - (float)h);
  }
}

// ------------------------------------------------------------- TDM load
// Load a 2D tile: 128 rows x 32 halfs (16 dwords) from global (row stride
// strideDw dwords) into LDS at byte offset ldsOff, padding +1 dword after
// every 16 dwords -> LDS row stride 34 halfs. D# per CDNA5 ISA.
__device__ __forceinline__ void tdm_tile_128x32(const _Float16* g,
                                                unsigned ldsOff,
                                                int strideDw) {
  unsigned long long ga = (unsigned long long)(uintptr_t)g;
  u32x4 g0;
  g0.x = 1u;                                   // count=1 (valid), no gather
  g0.y = ldsOff;                               // lds_addr [63:32]
  g0.z = (unsigned)ga;                         // global_addr [95:64]
  g0.w = (unsigned)((ga >> 32) & 0x1FFFFFFull) // global_addr [120:96]
       | (2u << 30);                           // type=2 (image)
  i32x8 g1;
  g1[0] = (2 << 16)       // data_size = 2 -> 4-byte elements
        | (1 << 20)       // pad_enable
        | (3 << 22);      // pad_interval = 3 -> every 16 dwords
                          // pad_amount = 0 -> 1 dword ; wg_mask = 0
  g1[1] = 0;                        // atomic_barrier=0 | tensor_dim0 lo16 (0)
  g1[2] = 0x4000;                   // tensor_dim0 hi16 (2^30: no OOB clip)
  g1[3] = 0x4000 | (16 << 16);      // tensor_dim1 hi16 | tile_dim0 = 16 dw
  g1[4] = 128;                      // tile_dim1 = 128 rows, tile_dim2 = 0
  g1[5] = strideDw;                 // tensor_dim0_stride lo32 (row stride)
  g1[6] = 0;                        // dim0_stride hi16 | dim1_stride lo16
  g1[7] = 0;                        // dim1_stride hi32 (unused for 2D)
  i32x4 z4;
  z4[0] = 0; z4[1] = 0; z4[2] = 0; z4[3] = 0;  // groups 2/3 unused (2D)
  i32x8 z8;
  z8[0] = 0; z8[1] = 0; z8[2] = 0; z8[3] = 0;
  z8[4] = 0; z8[5] = 0; z8[6] = 0; z8[7] = 0;
  __builtin_amdgcn_tensor_load_to_lds(g0, g1, z4, z4, z8, 0);
}

// ----------------------------------------------------------- WMMA GEMM
// C[M,N] = A[M,K] * B^T[N,K] (B given row-major [N][K] = original weight
// layout), split f16 inputs, f32 accumulate. Block tile 128x128, 256
// threads = 8 waves (2x4), wave tile 64x32 (4x2 WMMA accumulators).
// A/B tiles DMA'd by TDM into double-buffered padded LDS.
// EPI==0: store f32 C.  EPI==1: SiLU then re-split to f16 hi/lo.
template <int EPI>
__global__ __launch_bounds__(256)
void k_gemm_wmma(const _Float16* __restrict__ Ah, const _Float16* __restrict__ Al,
                 int lda,
                 const _Float16* __restrict__ Bh, const _Float16* __restrict__ Bl,
                 int ldb, int K,
                 float* __restrict__ Cf,
                 _Float16* __restrict__ Ch, _Float16* __restrict__ Cl, int ldc) {
  __shared__ __align__(16) _Float16 sAh[2][128][34];   // [m][k] padded
  __shared__ __align__(16) _Float16 sAl[2][128][34];
  __shared__ __align__(16) _Float16 sBh[2][128][34];   // [n][k] padded
  __shared__ __align__(16) _Float16 sBl[2][128][34];

  const int t    = threadIdx.x;
  const int lane = t & 31;
  const int wid  = t >> 5;
  const int wm   = (wid >> 2) * 64;   // wave M offset (2 wave-rows)
  const int wn   = (wid & 3) * 32;    // wave N offset (4 wave-cols)
  const int m0   = blockIdx.y * 128;
  const int n0   = blockIdx.x * 128;
  const int r8   = (lane & 16) ? 8 : 0;
  const int kb   = lane & 16;
  const int ln16 = lane & 15;

  v8f acc[4][2];
  #pragma unroll
  for (int i = 0; i < 4; i++)
    #pragma unroll
    for (int j = 0; j < 2; j++)
      #pragma unroll
      for (int r = 0; r < 8; r++) acc[i][j][r] = 0.0f;

  const int nIter = K >> 5;

  auto issue = [&](int buf, int k0) {
    tdm_tile_128x32(Ah + (size_t)m0 * lda + k0,
                    (unsigned)(uintptr_t)&sAh[buf][0][0], lda >> 1);
    tdm_tile_128x32(Al + (size_t)m0 * lda + k0,
                    (unsigned)(uintptr_t)&sAl[buf][0][0], lda >> 1);
    tdm_tile_128x32(Bh + (size_t)n0 * ldb + k0,
                    (unsigned)(uintptr_t)&sBh[buf][0][0], ldb >> 1);
    tdm_tile_128x32(Bl + (size_t)n0 * ldb + k0,
                    (unsigned)(uintptr_t)&sBl[buf][0][0], ldb >> 1);
  };

  // prologue: fill buffer 0
  if (wid == 0) {
    issue(0, 0);
    __builtin_amdgcn_s_wait_tensorcnt(0);
  }
  __syncthreads();

  for (int it = 0; it < nIter; ++it) {
    const int cur = it & 1;
    if (wid == 0 && it + 1 < nIter) issue(cur ^ 1, (it + 1) << 5);

    const _Float16 (*pAh)[34] = sAh[cur];
    const _Float16 (*pAl)[34] = sAl[cur];
    const _Float16 (*pBh)[34] = sBh[cur];
    const _Float16 (*pBl)[34] = sBl[cur];

    // ---- A fragments (row per lane, K = kb + 0..15)
    v16h ah[4], al[4], bh[2], bl[2];
    #pragma unroll
    for (int mt = 0; mt < 4; mt++) {
      int row = wm + mt * 16 + ln16;
      #pragma unroll
      for (int ep = 0; ep < 8; ep++) {
        int kk = kb + 2 * ep;
        ((uint32_t*)&ah[mt])[ep] = *(const uint32_t*)&pAh[row][kk];
        ((uint32_t*)&al[mt])[ep] = *(const uint32_t*)&pAl[row][kk];
      }
    }
    // ---- B fragments (col per lane, K = kb + 0..15)
    #pragma unroll
    for (int nt = 0; nt < 2; nt++) {
      int col = wn + nt * 16 + ln16;
      #pragma unroll
      for (int ep = 0; ep < 8; ep++) {
        ((uint32_t*)&bh[nt])[ep] = *(const uint32_t*)&pBh[col][kb + 2 * ep];
        ((uint32_t*)&bl[nt])[ep] = *(const uint32_t*)&pBl[col][kb + 2 * ep];
      }
    }
    // ---- split-precision WMMA: hi*hi + hi*lo + lo*hi
    #pragma unroll
    for (int mt = 0; mt < 4; mt++)
      #pragma unroll
      for (int nt = 0; nt < 2; nt++) {
        acc[mt][nt] = __builtin_amdgcn_wmma_f32_16x16x32_f16(
            false, ah[mt], false, bh[nt], (short)0, acc[mt][nt], false, false);
        acc[mt][nt] = __builtin_amdgcn_wmma_f32_16x16x32_f16(
            false, ah[mt], false, bl[nt], (short)0, acc[mt][nt], false, false);
        acc[mt][nt] = __builtin_amdgcn_wmma_f32_16x16x32_f16(
            false, al[mt], false, bh[nt], (short)0, acc[mt][nt], false, false);
      }

    // next buffer landed?  (issuing wave waits, everyone syncs)
    if (wid == 0 && it + 1 < nIter) __builtin_amdgcn_s_wait_tensorcnt(0);
    __syncthreads();
  }

  // ---- epilogue (C/D layout: VGPR r -> M = r + 8*lane[4], N = lane[3:0])
  #pragma unroll
  for (int mt = 0; mt < 4; mt++)
    #pragma unroll
    for (int nt = 0; nt < 2; nt++)
      #pragma unroll
      for (int r = 0; r < 8; r++) {
        int gm = m0 + wm + mt * 16 + r8 + r;
        int gn = n0 + wn + nt * 16 + ln16;
        float v = acc[mt][nt][r];
        if (EPI == 1) {
          float s = v / (1.0f + __expf(-v));   // SiLU
          _Float16 h = (_Float16)s;
          size_t o = (size_t)gm * ldc + gn;
          Ch[o] = h;
          Cl[o] = (_Float16)(s - (float)h);
        } else {
          Cf[(size_t)gm * ldc + gn] = v;
        }
      }
}

// ----------------------------------------------------------------- scan
// One workgroup (64 threads = 2 waves) per batch element. Thread i owns
// state row S[i][:] in 64 VGPRs. k,q broadcast via LDS.
__global__ __launch_bounds__(64)
void k_scan(const float* __restrict__ proj,      // [ROWS][256]: k|v|q|ax
            const float* __restrict__ da, const float* __restrict__ ba,
            _Float16* __restrict__ ch, _Float16* __restrict__ cl,
            float* __restrict__ Sfin) {
  const int b = blockIdx.x;
  const int i = threadIdx.x;
  __shared__ float kq[2][NST];
  float S[NST];
  #pragma unroll
  for (int j = 0; j < NST; j++) S[j] = 0.0f;
  const float dai = da[i], bai = ba[i];

  for (int t = 0; t < TT; t++) {
    const float* p = proj + ((size_t)b * TT + t) * 256;
    float kv = p[i], vv = p[64 + i], qv = p[128 + i], av = p[192 + i];
    kq[0][i] = kv;
    kq[1][i] = qv;
    __syncthreads();

    float ret = 0.0f;
    #pragma unroll
    for (int j = 0; j < NST; j++) ret += S[j] * kq[0][j];

    float alpha = 1.0f / (1.0f + __expf(-(av + dai * ret + bai)));
    float vn = (1.0f - alpha) * vv;

    float o = 0.0f;
    #pragma unroll
    for (int j = 0; j < NST; j++) {
      S[j] = alpha * S[j] + vn * kq[0][j];
      o += S[j] * kq[1][j];
    }
    float outv = o * o / (1.0f + __expf(-o));   // o * silu(o)

    size_t ci = ((size_t)b * TT + t) * NST + i;
    _Float16 h = (_Float16)outv;
    ch[ci] = h;
    cl[ci] = (_Float16)(outv - (float)h);
    __syncthreads();                            // protect LDS reuse
  }
  #pragma unroll
  for (int j = 0; j < NST; j++)
    Sfin[(size_t)b * NST * NST + (size_t)i * NST + j] = S[j];
}

// ------------------------------------------------------------- launcher
extern "C" void kernel_launch(void* const* d_in, const int* in_sizes, int n_in,
                              void* d_out, int out_size, void* d_ws,
                              size_t ws_size, hipStream_t stream) {
  (void)in_sizes; (void)n_in; (void)out_size; (void)ws_size;
  const float* x     = (const float*)d_in[0];
  const float* W_in  = (const float*)d_in[1];
  const float* W_k   = (const float*)d_in[2];
  const float* W_v   = (const float*)d_in[3];
  const float* W_q   = (const float*)d_in[4];
  const float* W_al  = (const float*)d_in[5];
  const float* d_al  = (const float*)d_in[6];
  const float* b_al  = (const float*)d_in[7];
  const float* W_out = (const float*)d_in[8];
  float* out = (float*)d_out;

  size_t off = 0;
  auto alloc = [&](size_t bytes) -> void* {
    off = (off + 255) & ~(size_t)255;
    void* p = (char*)d_ws + off;
    off += bytes;
    return p;
  };

  const size_t NX = (size_t)ROWS * DIM;
  _Float16* xh   = (_Float16*)alloc(NX * 2);
  _Float16* xl   = (_Float16*)alloc(NX * 2);
  _Float16* xph  = (_Float16*)alloc(NX * 2);
  _Float16* xpl  = (_Float16*)alloc(NX * 2);
  _Float16* Wih  = (_Float16*)alloc((size_t)DIM * DIM * 2);
  _Float16* Wil  = (_Float16*)alloc((size_t)DIM * DIM * 2);
  _Float16* Wph  = (_Float16*)alloc((size_t)256 * DIM * 2);
  _Float16* Wpl  = (_Float16*)alloc((size_t)256 * DIM * 2);
  _Float16* Woh  = (_Float16*)alloc((size_t)DIM * NST * 2);
  _Float16* Wol  = (_Float16*)alloc((size_t)DIM * NST * 2);
  float*    proj = (float*)   alloc((size_t)ROWS * 256 * 4);
  _Float16* ch   = (_Float16*)alloc((size_t)ROWS * NST * 2);
  _Float16* cl   = (_Float16*)alloc((size_t)ROWS * NST * 2);

  // 1) split activations and weights into hi/lo f16 (no transposes needed:
  //    original row-major [N][K] weight layout IS the B^T the GEMM wants)
  k_split<<<4096, 256, 0, stream>>>(x, xh, xl, NX);
  k_split<<<512, 256, 0, stream>>>(W_in, Wih, Wil, (size_t)DIM * DIM);
  k_split<<<64,  256, 0, stream>>>(W_k,  Wph,             Wpl,             (size_t)NST * DIM);
  k_split<<<64,  256, 0, stream>>>(W_v,  Wph +  64 * DIM, Wpl +  64 * DIM, (size_t)NST * DIM);
  k_split<<<64,  256, 0, stream>>>(W_q,  Wph + 128 * DIM, Wpl + 128 * DIM, (size_t)NST * DIM);
  k_split<<<64,  256, 0, stream>>>(W_al, Wph + 192 * DIM, Wpl + 192 * DIM, (size_t)NST * DIM);
  k_split<<<64,  256, 0, stream>>>(W_out, Woh, Wol, (size_t)DIM * NST);

  // 2) in_proj GEMM + fused SiLU, result re-split to f16 hi/lo
  k_gemm_wmma<1><<<dim3(DIM / 128, ROWS / 128), 256, 0, stream>>>(
      xh, xl, DIM, Wih, Wil, DIM, DIM, nullptr, xph, xpl, DIM);

  // 3) fused k|v|q|alpha projection GEMM -> f32 [ROWS][256]
  k_gemm_wmma<0><<<dim3(256 / 128, ROWS / 128), 256, 0, stream>>>(
      xph, xpl, DIM, Wph, Wpl, DIM, DIM, proj, nullptr, nullptr, 256);

  // 4) sequential gated scan; S_final goes to tail of d_out
  k_scan<<<BB, NST, 0, stream>>>(proj, d_al, b_al, ch, cl, out + NX);

  // 5) out_proj GEMM -> d_out [B,T,DIM]
  k_gemm_wmma<0><<<dim3(DIM / 128, ROWS / 128), 256, 0, stream>>>(
      ch, cl, NST, Woh, Wol, NST, NST, out, nullptr, nullptr, DIM);
}
